// BalancedTreeCell_41394894799306
// MI455X (gfx1250) — matrix operand order repaired and checked by reading, not compiled, not run on hardware
//
#include <hip/hip_runtime.h>
#include <hip/hip_bf16.h>

// ---------------------------------------------------------------------------
// BalancedTreeCell for MI455X (gfx1250, wave32, WMMA bf16)
//   - bf16 WMMA 16x16x32, f32 accumulate (compute-bound workload)
//   - weights pre-swizzled to B-fragment order (coalesced global_load_b128)
//   - 32-row M tiles: each B fragment fetched from L2 feeds 2 WMMAs,
//     halving L2 weight traffic vs 16-row tiles (the binding resource)
// ---------------------------------------------------------------------------
typedef __attribute__((ext_vector_type(16))) __bf16 v16bf;
typedef __attribute__((ext_vector_type(8)))  __bf16 v8bf;
typedef __attribute__((ext_vector_type(8)))  float  v8f;

#define Dm      256          // model dim
#define TWO_D   512
#define Hh      1024         // hidden dim
#define NBATCH  16
#define S0      4096
#define TMe     16           // rows per block, embed kernel
#define TMc     32           // rows per block, cell kernel
#define THREADS_E 256        // 8 waves
#define THREADS_C 512        // 16 waves
#define LNEPS   1e-5f

__device__ __forceinline__ unsigned short f2bf(float f) {
    unsigned int u = __float_as_uint(f);
    u += 0x7FFFu + ((u >> 16) & 1u);          // round-to-nearest-even
    return (unsigned short)(u >> 16);
}
__device__ __forceinline__ float gelu_exact(float x) {
    return 0.5f * x * (1.0f + erff(x * 0.70710678118654752f));
}
__device__ __forceinline__ float sigmoidf_fast(float x) {
    return 1.0f / (1.0f + __expf(-x));
}

// Load a 16x32 bf16 A-fragment from LDS (ISA layout: lane half selects K base).
__device__ __forceinline__ v16bf load_a_frag(const unsigned short* sA, int strideA,
                                             int lhalf, int half, int k0) {
    const unsigned short* ap = sA + lhalf * strideA + k0 + half * 8;
    v8bf alo = *(const v8bf*)(ap);        // K = k0 + half*8 + [0..7]
    v8bf ahi = *(const v8bf*)(ap + 16);   // K = k0 + 16 + half*8 + [0..7]
    return __builtin_shufflevector(alo, ahi,
                                   0,1,2,3,4,5,6,7,8,9,10,11,12,13,14,15);
}

// ---------------------------------------------------------------------------
// fp32 weight -> bf16, pre-swizzled into WMMA B-fragment tile order:
// tile (kt,nt) holds 32 lanes x 16 contiguous bf16 each.
// ---------------------------------------------------------------------------
__global__ void swizzle_w_bf16(const float* __restrict__ W,
                               unsigned short* __restrict__ out,
                               int K, int N) {
    int idx = blockIdx.x * blockDim.x + threadIdx.x;
    int total = K * N;
    if (idx >= total) return;
    int k = idx / N, n = idx % N;
    int kt = k >> 5, kr = k & 31, half = (kr >> 3) & 1, e = ((kr >> 4) << 3) | (kr & 7);
    int nt = n >> 4, nl = n & 15;
    int ntiles_n = N >> 4;
    size_t o = (((size_t)(kt * ntiles_n + nt) * 32) + (half * 16 + nl)) * 16 + e;
    out[o] = f2bf(W[idx]);
}

// ---------------------------------------------------------------------------
// Embedding: state = LN(input @ w_word + b_word)   rows = 16*4096
// ---------------------------------------------------------------------------
__global__ __launch_bounds__(THREADS_E)
void embed_ln_kernel(const float* __restrict__ x,
                     const unsigned short* __restrict__ wsw,   // swizzled 256x256
                     const float* __restrict__ bw,
                     const float* __restrict__ g,
                     const float* __restrict__ bb,
                     float* __restrict__ out) {
    const int SA = Dm + 8;          // bf16 elements, 16B-aligned rows
    const int SC = Dm + 16;
    __shared__ __align__(16) unsigned short sA[TMe * SA];
    __shared__ float sC[TMe * SC];

    int t = threadIdx.x;
    int row0 = blockIdx.x * TMe;

    for (int i = t; i < TMe * Dm; i += THREADS_E) {
        int r = i / Dm, c = i % Dm;
        sA[r * SA + c] = f2bf(x[(size_t)(row0 + r) * Dm + c]);
    }
    __syncthreads();

    int wid = t >> 5, lane = t & 31, half = lane >> 4, lhalf = lane & 15;
    const int ntn = Dm / 16;        // 16 n-tiles, 2 per wave
    v8f acc0 = {}, acc1 = {};
    for (int ks = 0; ks < Dm / 32; ++ks) {
        v16bf a = load_a_frag(sA, SA, lhalf, half, ks * 32);
        const unsigned short* bbase =
            wsw + ((size_t)(ks * ntn + wid * 2) * 32 + lane) * 16;
        v16bf b0 = *(const v16bf*)(bbase);
        v16bf b1 = *(const v16bf*)(bbase + 32 * 16);
        acc0 = __builtin_amdgcn_wmma_f32_16x16x32_bf16(false, a, false, b0,
                                                       (short)0, acc0, false, false);
        acc1 = __builtin_amdgcn_wmma_f32_16x16x32_bf16(false, a, false, b1,
                                                       (short)0, acc1, false, false);
    }
#pragma unroll
    for (int r = 0; r < 8; ++r) {
        int m = r + half * 8;
        int c0 = wid * 32 + lhalf;
        sC[m * SC + c0]      = acc0[r] + bw[c0];
        sC[m * SC + c0 + 16] = acc1[r] + bw[c0 + 16];
    }
    __syncthreads();

    // LayerNorm: 16 lanes per row, 16 cols per lane
    int rrow = t >> 4, c16 = t & 15;
    float y[16]; float s = 0.f, ss = 0.f;
#pragma unroll
    for (int q = 0; q < 16; ++q) {
        float v = sC[rrow * SC + (c16 * 16 + q)];
        y[q] = v; s += v; ss += v * v;
    }
#pragma unroll
    for (int m = 8; m >= 1; m >>= 1) {
        s  += __shfl_xor(s,  m, 16);
        ss += __shfl_xor(ss, m, 16);
    }
    float mu   = s * (1.0f / Dm);
    float var  = ss * (1.0f / Dm) - mu * mu;
    float rstd = rsqrtf(var + LNEPS);
    size_t ob = (size_t)(row0 + rrow) * Dm;
#pragma unroll
    for (int q = 0; q < 16; ++q) {
        int dcol = c16 * 16 + q;
        out[ob + dcol] = (y[q] - mu) * rstd * g[dcol] + bb[dcol];
    }
}

// ---------------------------------------------------------------------------
// One tree merge level, fully fused, 32 rows (pairs) per 512-thread block:
//   h = gelu([l|r] @ w1 + b1); c = h @ w2 + b2
//   y = LN(sig(c0)*l + sig(c1)*r + sig(c2)*c3); out = mr*y + (1-mr)*l
// Each wave owns a 32x64 output slice (2 M-tiles x 4 N-tiles); every B
// fragment loaded from L2 is reused by both M-tiles.
// ---------------------------------------------------------------------------
__global__ __launch_bounds__(THREADS_C)
void tree_cell_kernel(const float* __restrict__ state_in,   // [NB][Scur][D]
                      float* __restrict__ state_out,          // [NB][Sh][D]
                      const float* __restrict__ mask,         // [NB][S0]
                      const unsigned short* __restrict__ w1s, // swz 512x1024
                      const float* __restrict__ b1,
                      const unsigned short* __restrict__ w2s, // swz 1024x1024
                      const float* __restrict__ b2,
                      const float* __restrict__ lg,
                      const float* __restrict__ lb,
                      int Scur, int level) {
    const int Sh  = Scur >> 1;
    const int SA  = TWO_D + 8;     // bf16 elems
    const int SHs = Hh + 8;        // bf16 elems
    const int SC  = Hh + 16;       // fp32 elems
    extern __shared__ __align__(16) unsigned char smem[];
    unsigned short* sA = (unsigned short*)smem;                          // 33.3 KB
    unsigned short* sH = (unsigned short*)(smem + TMc * SA * 2);         // 66.0 KB
    float*          sC = (float*)(smem + TMc * SA * 2 + TMc * SHs * 2);  // 133.1 KB

    int t  = threadIdx.x;
    int p0 = blockIdx.x * TMc;
    int total = NBATCH * Sh;

    // stage cat = [l | r] as bf16; inactive rows zeroed (EXEC stays full for WMMA)
    for (int i = t; i < TMc * TWO_D; i += THREADS_C) {
        int r = i / TWO_D, c = i % TWO_D;
        int p = p0 + r;
        float v = 0.0f;
        if (p < total) {
            int bn = p / Sh, j = p % Sh;
            size_t base = ((size_t)bn * Scur + 2 * j) * Dm;
            v = state_in[base + c];         // c<256 -> l, else r (contiguous)
        }
        sA[r * SA + c] = f2bf(v);
    }
    __syncthreads();

    int wid = t >> 5, lane = t & 31, half = lane >> 4, lhalf = lane & 15;
    const int ntn = Hh / 16;                // 64 n-tiles; 4 per wave (16 waves)

    // ---------------- GEMM1: 32 x 512 x 1024, gelu -> sH (bf16) -------------
    {
        v8f acc[2][4] = {};
        for (int ks = 0; ks < TWO_D / 32; ++ks) {
            v16bf a0 = load_a_frag(sA,           SA, lhalf, half, ks * 32);
            v16bf a1 = load_a_frag(sA + 16 * SA, SA, lhalf, half, ks * 32);
            const unsigned short* bbase =
                w1s + ((size_t)(ks * ntn + wid * 4) * 32 + lane) * 16;
            if (ks + 1 < TWO_D / 32)
                __builtin_prefetch((const void*)(bbase + (size_t)ntn * 32 * 16), 0, 1);
#pragma unroll
            for (int nt = 0; nt < 4; ++nt) {
                v16bf b = *(const v16bf*)(bbase + nt * (32 * 16));
                acc[0][nt] = __builtin_amdgcn_wmma_f32_16x16x32_bf16(
                    false, a0, false, b, (short)0, acc[0][nt], false, false);
                acc[1][nt] = __builtin_amdgcn_wmma_f32_16x16x32_bf16(
                    false, a1, false, b, (short)0, acc[1][nt], false, false);
            }
        }
#pragma unroll
        for (int nt = 0; nt < 4; ++nt) {
            int col = wid * 64 + nt * 16 + lhalf;
            float bv = b1[col];
#pragma unroll
            for (int mt = 0; mt < 2; ++mt) {
#pragma unroll
                for (int r = 0; r < 8; ++r) {
                    int m = mt * 16 + r + half * 8;
                    sH[m * SHs + col] = f2bf(gelu_exact(acc[mt][nt][r] + bv));
                }
            }
        }
    }
    __syncthreads();

    // ---------------- GEMM2: 32 x 1024 x 1024 -> sC (fp32) ------------------
    {
        v8f acc[2][4] = {};
        for (int ks = 0; ks < Hh / 32; ++ks) {
            v16bf a0 = load_a_frag(sH,            SHs, lhalf, half, ks * 32);
            v16bf a1 = load_a_frag(sH + 16 * SHs, SHs, lhalf, half, ks * 32);
            const unsigned short* bbase =
                w2s + ((size_t)(ks * ntn + wid * 4) * 32 + lane) * 16;
            if (ks + 1 < Hh / 32)
                __builtin_prefetch((const void*)(bbase + (size_t)ntn * 32 * 16), 0, 1);
#pragma unroll
            for (int nt = 0; nt < 4; ++nt) {
                v16bf b = *(const v16bf*)(bbase + nt * (32 * 16));
                acc[0][nt] = __builtin_amdgcn_wmma_f32_16x16x32_bf16(
                    false, a0, false, b, (short)0, acc[0][nt], false, false);
                acc[1][nt] = __builtin_amdgcn_wmma_f32_16x16x32_bf16(
                    false, a1, false, b, (short)0, acc[1][nt], false, false);
            }
        }
#pragma unroll
        for (int nt = 0; nt < 4; ++nt) {
            int col = wid * 64 + nt * 16 + lhalf;
            float bv = b2[col];
#pragma unroll
            for (int mt = 0; mt < 2; ++mt) {
#pragma unroll
                for (int r = 0; r < 8; ++r) {
                    int m = mt * 16 + r + half * 8;
                    sC[m * SC + col] = acc[mt][nt][r] + bv;
                }
            }
        }
    }
    __syncthreads();

    // ------------- gates + gated sum + LayerNorm + mask blend ---------------
    int rrow = t >> 4, c16 = t & 15;       // 512 threads = 32 rows x 16 lanes
    int p = p0 + rrow;
    bool active = (p < total);
    int bn = active ? (p / Sh) : 0;
    int j  = active ? (p % Sh) : 0;
    size_t lrbase = ((size_t)bn * Scur + 2 * j) * Dm;

    float y[16], lsv[16];
    float s = 0.f, ss = 0.f;
#pragma unroll
    for (int q = 0; q < 16; ++q) {
        int dcol = c16 * 16 + q;
        float f1 = sigmoidf_fast(sC[rrow * SC + dcol]);
        float f2 = sigmoidf_fast(sC[rrow * SC + Dm + dcol]);
        float gi = sigmoidf_fast(sC[rrow * SC + 2 * Dm + dcol]);
        float pa = sC[rrow * SC + 3 * Dm + dcol];
        float lv = active ? state_in[lrbase + dcol] : 0.0f;
        float rv = active ? state_in[lrbase + Dm + dcol] : 0.0f;
        float v  = f1 * lv + f2 * rv + gi * pa;
        y[q] = v; lsv[q] = lv;
        s += v; ss += v * v;
    }
#pragma unroll
    for (int m = 8; m >= 1; m >>= 1) {
        s  += __shfl_xor(s,  m, 16);
        ss += __shfl_xor(ss, m, 16);
    }
    float mu   = s * (1.0f / Dm);
    float var  = ss * (1.0f / Dm) - mu * mu;
    float rstd = rsqrtf(var + LNEPS);

    float mr = active ? mask[(size_t)bn * S0 + (((2 * j + 1) << level))] : 0.0f;
    size_t ob = ((size_t)bn * Sh + j) * Dm;
#pragma unroll
    for (int q = 0; q < 16; ++q) {
        int dcol = c16 * 16 + q;
        float o = (y[q] - mu) * rstd * lg[dcol] + lb[dcol];
        float res = mr * o + (1.0f - mr) * lsv[q];
        if (active) state_out[ob + dcol] = res;
    }
}

// ---------------------------------------------------------------------------
extern "C" void kernel_launch(void* const* d_in, const int* in_sizes, int n_in,
                              void* d_out, int out_size, void* d_ws, size_t ws_size,
                              hipStream_t stream) {
    (void)in_sizes; (void)n_in; (void)out_size; (void)ws_size;
    const float* input  = (const float*)d_in[0];
    const float* mask   = (const float*)d_in[1];
    const float* w_word = (const float*)d_in[2];
    const float* b_word = (const float*)d_in[3];
    const float* w1     = (const float*)d_in[4];
    const float* bias1  = (const float*)d_in[5];
    const float* w2     = (const float*)d_in[6];
    const float* bias2  = (const float*)d_in[7];
    const float* ln0_g  = (const float*)d_in[8];
    const float* ln0_b  = (const float*)d_in[9];
    const float* lnc_g  = (const float*)d_in[10];
    const float* lnc_b  = (const float*)d_in[11];

    char* ws = (char*)d_ws;
    unsigned short* wswWord = (unsigned short*)ws; ws += (size_t)Dm * Dm * 2;     // 128 KB
    unsigned short* wsw1    = (unsigned short*)ws; ws += (size_t)TWO_D * Hh * 2;  // 1 MB
    unsigned short* wsw2    = (unsigned short*)ws; ws += (size_t)Hh * Hh * 2;     // 2 MB
    float* stateA = (float*)ws; ws += (size_t)NBATCH * S0 * Dm * sizeof(float);   // 64 MB
    float* stateB = (float*)ws;                                                   // 32 MB

    const int thr = 256;
    swizzle_w_bf16<<<(Dm * Dm + thr - 1) / thr, thr, 0, stream>>>(w_word, wswWord, Dm, Dm);
    swizzle_w_bf16<<<(TWO_D * Hh + thr - 1) / thr, thr, 0, stream>>>(w1, wsw1, TWO_D, Hh);
    swizzle_w_bf16<<<(Hh * Hh + thr - 1) / thr, thr, 0, stream>>>(w2, wsw2, Hh, Hh);

    // embedding + LN
    embed_ln_kernel<<<(NBATCH * S0) / TMe, THREADS_E, 0, stream>>>(
        input, wswWord, b_word, ln0_g, ln0_b, stateA);

    // 12 dependent merge levels, ping-pong state buffers; last level -> d_out
    size_t cellSmem = (size_t)TMc * (TWO_D + 8) * 2   // sA  (33.3 KB)
                    + (size_t)TMc * (Hh + 8) * 2       // sH  (66.0 KB)
                    + (size_t)TMc * (Hh + 16) * 4;     // sC  (133.1 KB) -> 232 KB of 320 KB
    float* bufs[2] = { stateA, stateB };
    int Scur = S0, level = 0;
    while (Scur > 1) {
        int Sh = Scur >> 1;
        float* in  = bufs[level & 1];
        float* out = (Sh == 1) ? (float*)d_out : bufs[(level + 1) & 1];
        int total  = NBATCH * Sh;
        int blocks = (total + TMc - 1) / TMc;
        tree_cell_kernel<<<blocks, THREADS_C, cellSmem, stream>>>(
            in, out, mask, wsw1, bias1, wsw2, bias2, lnc_g, lnc_b, Scur, level);
        Scur = Sh; ++level;
    }
}
